// GMMEncoder_1391569404522
// MI455X (gfx1250) — matrix operand order repaired and checked by reading, not compiled, not run on hardware
//
#include <hip/hip_runtime.h>
#include <hip/hip_bf16.h>

// ---------------------------------------------------------------------------
// CDNA5 (gfx1250) implementation of the GAT -> pool -> biLSTM -> GMM encoder.
// Dense GEMMs use v_wmma_f32_16x16x32_f16; weights pre-transposed to [N][K]
// so every WMMA fragment load is contiguous (global_load_b128 / ds_load).
// LSTM gate weights live in LDS (216KB of the 320KB WGP LDS) across steps.
// ---------------------------------------------------------------------------

typedef __attribute__((ext_vector_type(16))) _Float16 v16h;
typedef __attribute__((ext_vector_type(8)))  float    v8f;

#define B_  8
#define T_  64
#define N_  64
#define F_  32
#define HID 64
#define HDS 4
#define RNN 128
#define G_  (B_ * T_)        // 512 graphs
#define NTOT (G_ * N_)       // 32768 nodes
#define EBASE 512
#define EMAX  (EBASE + N_)   // 576 edges incl self loops (shared topology)

// ---------------- WMMA tile loaders (ISA 7.12.2 layouts, wave32) ------------

// A matrix 16x32 f16, row-major source, all 16 rows valid.
__device__ inline v16h load_a16(const _Float16* __restrict__ A, int ld, int k0, int lane) {
    int row = lane & 15;
    int hi  = lane >> 4;           // lanes 16-31 hold K+8 / K+24 groups
    const _Float16* p = A + (size_t)row * ld + k0 + hi * 8;
    v16h a;
#pragma unroll
    for (int e = 0; e < 8; ++e) { a[e] = p[e]; a[8 + e] = p[16 + e]; }
    return a;
}

// A matrix with only `valid` rows (zero-pad the rest) — for M=8 LSTM batch.
__device__ inline v16h load_a16_rows(const _Float16* __restrict__ A, int ld, int k0,
                                     int lane, int valid) {
    int row = lane & 15;
    int hi  = lane >> 4;
    v16h a;
    if (row < valid) {
        const _Float16* p = A + (size_t)row * ld + k0 + hi * 8;
#pragma unroll
        for (int e = 0; e < 8; ++e) { a[e] = p[e]; a[8 + e] = p[16 + e]; }
    } else {
#pragma unroll
        for (int e = 0; e < 16; ++e) a[e] = (_Float16)0.0f;
    }
    return a;
}

// B matrix 32x16 built as W^T from row-major W[N][K]: two 16-byte runs/lane (k+0..7, k+16..23).
__device__ inline v16h load_b16_T(const _Float16* __restrict__ W, int ldk, int n0, int k0,
                                  int lane) {
    int n  = n0 + (lane & 15);
    int hi = lane >> 4;            // lanes 16-31 hold K+8 / K+24 groups
    const _Float16* p = W + (size_t)n * ldk + k0 + hi * 8;
    v16h b;
#pragma unroll
    for (int e = 0; e < 8; ++e) { b[e] = p[e]; b[8 + e] = p[16 + e]; }
    return b;
}

// C/D 16x16 f32: VGPR r holds row r (lanes 0-15) / r+8 (lanes 16-31).
__device__ inline void store_c16(float* __restrict__ C, int ld, int m0, int n0, int lane,
                                 v8f acc) {
    int hi  = lane >> 4;
    int col = n0 + (lane & 15);
#pragma unroll
    for (int r = 0; r < 8; ++r) C[(size_t)(m0 + r + hi * 8) * ld + col] = acc[r];
}

// ---------------- f16 WMMA GEMM: C[MxN] = A[MxK] * BT[NxK]^T ----------------
// One wave per block; each wave computes a 16x64 strip: A fragment reused
// across 4 back-to-back v_wmma per k-step.

__global__ void wmma_gemm_f16T(const _Float16* __restrict__ A,
                               const _Float16* __restrict__ BT,
                               float* __restrict__ C, int M, int N, int K) {
    int lane = threadIdx.x & 31;
    int n0 = blockIdx.x * 64;
    int m0 = blockIdx.y * 16;
    const _Float16* Arow = A + (size_t)m0 * K;
    v8f acc0 = {}, acc1 = {}, acc2 = {}, acc3 = {};
    for (int k0 = 0; k0 < K; k0 += 32) {
        if (k0 + 32 < K) __builtin_prefetch(Arow + k0 + 32, 0, 0);   // global_prefetch_b8
        v16h a  = load_a16(Arow, K, k0, lane);
        v16h b0 = load_b16_T(BT, K, n0,      k0, lane);
        v16h b1 = load_b16_T(BT, K, n0 + 16, k0, lane);
        v16h b2 = load_b16_T(BT, K, n0 + 32, k0, lane);
        v16h b3 = load_b16_T(BT, K, n0 + 48, k0, lane);
        acc0 = __builtin_amdgcn_wmma_f32_16x16x32_f16(false, a, false, b0, (short)0, acc0, false, false);
        acc1 = __builtin_amdgcn_wmma_f32_16x16x32_f16(false, a, false, b1, (short)0, acc1, false, false);
        acc2 = __builtin_amdgcn_wmma_f32_16x16x32_f16(false, a, false, b2, (short)0, acc2, false, false);
        acc3 = __builtin_amdgcn_wmma_f32_16x16x32_f16(false, a, false, b3, (short)0, acc3, false, false);
    }
    store_c16(C, N, m0, n0,      lane, acc0);
    store_c16(C, N, m0, n0 + 16, lane, acc1);
    store_c16(C, N, m0, n0 + 32, lane, acc2);
    store_c16(C, N, m0, n0 + 48, lane, acc3);
}

// ---------------- small helpers --------------------------------------------

__global__ void f32_to_f16(const float* __restrict__ s, _Float16* __restrict__ d, int n) {
    int i = blockIdx.x * blockDim.x + threadIdx.x;
    if (i < n) d[i] = (_Float16)s[i];
}

// W [K][N] row-major (f32)  ->  WT [N][K] (f16)
__global__ void transpose_to_f16(const float* __restrict__ W, _Float16* __restrict__ WT,
                                 int K, int N) {
    int idx = blockIdx.x * blockDim.x + threadIdx.x;
    if (idx >= K * N) return;
    int k = idx / N, n = idx % N;
    WT[(size_t)n * K + k] = (_Float16)W[idx];
}

__device__ inline float lrelu(float x) { return x > 0.0f ? x : 0.2f * x; }
__device__ inline float sigf(float x)  { return 1.0f / (1.0f + __expf(-x)); }

// ---------------- CSR of the shared per-graph topology (incl self loops) ----

__global__ void build_csr(const int* __restrict__ ei, int* __restrict__ off,
                          int* __restrict__ list) {
    __shared__ int s_deg[N_];
    __shared__ int s_off[N_ + 1];
    int d = threadIdx.x;                       // 64 threads
    int cnt = 1;                               // self loop
    for (int e = 0; e < EBASE; ++e) cnt += (ei[EBASE + e] == d) ? 1 : 0;
    s_deg[d] = cnt;
    __syncthreads();
    if (d == 0) {
        int acc = 0;
        for (int i = 0; i < N_; ++i) { s_off[i] = acc; acc += s_deg[i]; }
        s_off[N_] = acc;
    }
    __syncthreads();
    int pos = s_off[d];
    for (int e = 0; e < EBASE; ++e)
        if (ei[EBASE + e] == d) list[pos++] = ei[e];
    list[pos++] = d;                           // self loop
    off[d] = s_off[d];
    if (d == 0) off[N_] = s_off[N_];
}

// ---------------- GAT attention scores per (node, head) ---------------------

template <int H, int C>
__global__ void gat_scores(const float* __restrict__ Hb, const float* __restrict__ a_s,
                           const float* __restrict__ a_d, float* __restrict__ AS,
                           float* __restrict__ AD) {
    int idx = blockIdx.x * blockDim.x + threadIdx.x;
    if (idx >= NTOT * H) return;
    int node = idx / H, h = idx % H;
    const float* row = Hb + (size_t)node * (H * C) + h * C;
    float s = 0.0f, dd = 0.0f;
#pragma unroll 4
    for (int c = 0; c < C; ++c) { s += row[c] * a_s[h * C + c]; dd += row[c] * a_d[h * C + c]; }
    AS[idx] = s;
    AD[idx] = dd;
}

// ---------------- per-graph GAT softmax + aggregation (no atomics) ----------
// The graph's 64 x HC h-tile is staged once in dynamic LDS; phase-B gathers
// hit LDS instead of re-reading global ~9x.

template <int H, int C>
__global__ void gat_aggregate(const float* __restrict__ Hb, const float* __restrict__ AS,
                              const float* __restrict__ AD, const int* __restrict__ off,
                              const int* __restrict__ list, const float* __restrict__ bias,
                              float* __restrict__ Xout, _Float16* __restrict__ X16) {
    const int HC = H * C;                      // == blockDim.x
    extern __shared__ float s_h[];             // [N_ * HC] staged h-tile
    __shared__ float s_as[N_ * H], s_ad[N_ * H], s_mx[N_ * H], s_sm[N_ * H];
    __shared__ int   s_off[N_ + 1], s_list[EMAX];
    int g = blockIdx.x, tid = threadIdx.x;
    size_t base = (size_t)g * N_;
    for (int i = tid; i < N_ * HC; i += HC) s_h[i] = Hb[base * HC + i];
    for (int i = tid; i < N_ * H; i += HC) { s_as[i] = AS[base * H + i]; s_ad[i] = AD[base * H + i]; }
    for (int i = tid; i < N_ + 1; i += HC) s_off[i] = off[i];
    for (int i = tid; i < EMAX; i += HC) s_list[i] = list[i];
    __syncthreads();
    // phase A: per (dst,head) running max then exp-sum
    for (int p = tid; p < N_ * H; p += HC) {
        int d = p / H, h = p % H;
        float ad = s_ad[d * H + h];
        float mx = -1e30f;
        for (int j = s_off[d]; j < s_off[d + 1]; ++j)
            mx = fmaxf(mx, lrelu(s_as[s_list[j] * H + h] + ad));
        float sm = 0.0f;
        for (int j = s_off[d]; j < s_off[d + 1]; ++j)
            sm += __expf(lrelu(s_as[s_list[j] * H + h] + ad) - mx);
        s_mx[p] = mx; s_sm[p] = sm;
    }
    __syncthreads();
    // phase B: each thread owns one output channel, gather over incoming edges
    int col = tid, h = col / C;
    for (int d = 0; d < N_; ++d) {
        float ad = s_ad[d * H + h];
        float mx = s_mx[d * H + h];
        float inv = 1.0f / s_sm[d * H + h];
        float acc = 0.0f;
        for (int j = s_off[d]; j < s_off[d + 1]; ++j) {
            int src = s_list[j];
            float w = __expf(lrelu(s_as[src * H + h] + ad) - mx) * inv;
            acc += w * s_h[src * HC + col];
        }
        float v = acc + bias[col];
        v = v > 0.0f ? v : 0.0f;               // layer ReLU
        Xout[(base + d) * HC + col] = v;
        X16[(base + d) * HC + col]  = (_Float16)v;
    }
}

// ---------------- attention pooling per graph + mask ------------------------

__global__ void pool_kernel(const float* __restrict__ X, const float* __restrict__ pw,
                            const float* __restrict__ pb, const float* __restrict__ mask,
                            _Float16* __restrict__ ge16) {
    __shared__ float s_l[N_];
    __shared__ float s_red[2];
    int g = blockIdx.x, n = threadIdx.x;       // 64 threads
    const float* row = X + ((size_t)g * N_ + n) * HID;
    float l = pb[0];
#pragma unroll 4
    for (int c = 0; c < HID; ++c) l += row[c] * pw[c];
    s_l[n] = l;
    __syncthreads();
    if (n == 0) { float m = s_l[0]; for (int i = 1; i < N_; ++i) m = fmaxf(m, s_l[i]); s_red[0] = m; }
    __syncthreads();
    s_l[n] = __expf(l - s_red[0]);
    __syncthreads();
    if (n == 0) { float s = 0.0f; for (int i = 0; i < N_; ++i) s += s_l[i]; s_red[1] = s; }
    __syncthreads();
    float inv = 1.0f / s_red[1];
    float mk = mask[g];
    float acc = 0.0f;                          // thread owns channel n
    for (int j = 0; j < N_; ++j) acc += s_l[j] * X[((size_t)g * N_ + j) * HID + n];
    ge16[(size_t)g * HID + n] = (_Float16)(acc * inv * mk);
}

// ---------------- bidirectional LSTM (1 block per direction) ----------------
// Gate weights staged once in 216KB of LDS (CDNA5: 320KB/WGP); the serial
// 64-step chain feeds WMMA fragments from ds_load instead of L2.

__global__ void lstm_kernel(const _Float16* __restrict__ ge16,
                            const _Float16* __restrict__ WihF, const _Float16* __restrict__ WhhF,
                            const float* __restrict__ bihF, const float* __restrict__ bhhF,
                            const _Float16* __restrict__ WihR, const _Float16* __restrict__ WhhR,
                            const float* __restrict__ bihR, const float* __restrict__ bhhR,
                            float* __restrict__ tf) {
    extern __shared__ char smem[];
    _Float16* s_wih = (_Float16*)smem;                 // [4*RNN][HID]  64KB
    _Float16* s_whh = s_wih + 4 * RNN * HID;           // [4*RNN][RNN] 128KB
    _Float16* s_h16 = s_whh + 4 * RNN * RNN;           // [16][RNN]      4KB (rows 8..15 zero)
    float*    s_c   = (float*)(s_h16 + 16 * RNN);      // [8][RNN]       4KB
    float*    s_g   = s_c + B_ * RNN;                  // [8][4*RNN]    16KB
    int dir = blockIdx.x;
    const _Float16* Wih = dir ? WihR : WihF;
    const _Float16* Whh = dir ? WhhR : WhhF;
    const float* bih = dir ? bihR : bihF;
    const float* bhh = dir ? bhhR : bhhF;
    int tid = threadIdx.x;                     // 512 threads = 16 waves
    int wave = tid >> 5, lane = tid & 31;
    for (int i = tid; i < 4 * RNN * HID; i += 512) s_wih[i] = Wih[i];
    for (int i = tid; i < 4 * RNN * RNN; i += 512) s_whh[i] = Whh[i];
    for (int i = tid; i < 16 * RNN; i += 512) s_h16[i] = (_Float16)0.0f;
    for (int i = tid; i < B_ * RNN; i += 512) s_c[i] = 0.0f;
    __syncthreads();
    for (int t = 0; t < T_; ++t) {
        int tau = dir ? (T_ - 1 - t) : t;
        const _Float16* Ax = ge16 + (size_t)tau * HID;   // row b -> + b*T_*HID
#pragma unroll
        for (int sub = 0; sub < 2; ++sub) {
            int n0 = wave * 32 + sub * 16;
            float bsum = bih[n0 + (lane & 15)] + bhh[n0 + (lane & 15)];
            v8f acc;
#pragma unroll
            for (int r = 0; r < 8; ++r) acc[r] = bsum;
            // x_t @ Wih^T  (K = HID = 64), weights from LDS
#pragma unroll
            for (int k0 = 0; k0 < HID; k0 += 32) {
                v16h a = load_a16_rows(Ax, T_ * HID, k0, lane, B_);
                v16h b = load_b16_T(s_wih, HID, n0, k0, lane);
                acc = __builtin_amdgcn_wmma_f32_16x16x32_f16(false, a, false, b,
                                                             (short)0, acc, false, false);
            }
            // h @ Whh^T  (K = RNN = 128), state + weights from LDS
#pragma unroll
            for (int k0 = 0; k0 < RNN; k0 += 32) {
                v16h a = load_a16(s_h16, RNN, k0, lane);
                v16h b = load_b16_T(s_whh, RNN, n0, k0, lane);
                acc = __builtin_amdgcn_wmma_f32_16x16x32_f16(false, a, false, b,
                                                             (short)0, acc, false, false);
            }
            int hi = lane >> 4, col = n0 + (lane & 15);
#pragma unroll
            for (int r = 0; r < 8; ++r) {
                int m = r + hi * 8;
                if (m < B_) s_g[m * (4 * RNN) + col] = acc[r];
            }
        }
        __syncthreads();
        for (int idx = tid; idx < B_ * RNN; idx += 512) {
            int b = idx >> 7, j = idx & (RNN - 1);
            const float* gr = s_g + b * (4 * RNN);
            float gi = gr[j], gf = gr[RNN + j], gg = gr[2 * RNN + j], go = gr[3 * RNN + j];
            float c = sigf(gf) * s_c[idx] + sigf(gi) * tanhf(gg);
            s_c[idx] = c;
            float h = sigf(go) * tanhf(c);
            s_h16[b * RNN + j] = (_Float16)h;
            if (t == T_ - 1) tf[b * (2 * RNN) + dir * RNN + j] = h;
        }
        __syncthreads();
    }
}

// ---------------- GMM output heads -----------------------------------------

__global__ void head_kernel(const float* __restrict__ tf, const float* __restrict__ W,
                            const float* __restrict__ bias, float* __restrict__ out,
                            int n_out) {
    int idx = blockIdx.x * blockDim.x + threadIdx.x;
    if (idx >= B_ * n_out) return;
    int b = idx / n_out, n = idx % n_out;
    float acc = bias[n];
#pragma unroll 4
    for (int k = 0; k < 2 * RNN; ++k) acc += tf[b * (2 * RNN) + k] * W[(size_t)k * n_out + n];
    out[idx] = acc;
}

// ---------------------------------------------------------------------------

extern "C" void kernel_launch(void* const* d_in, const int* in_sizes, int n_in,
                              void* d_out, int out_size, void* d_ws, size_t ws_size,
                              hipStream_t stream) {
    (void)in_sizes; (void)n_in; (void)out_size; (void)ws_size;
    const float* x      = (const float*)d_in[0];
    const int*   ei     = (const int*)d_in[1];
    const float* mask   = (const float*)d_in[2];
    const float* W0     = (const float*)d_in[3];
    const float* as0    = (const float*)d_in[4];
    const float* ad0    = (const float*)d_in[5];
    const float* b0     = (const float*)d_in[6];
    const float* W1     = (const float*)d_in[7];
    const float* as1    = (const float*)d_in[8];
    const float* ad1    = (const float*)d_in[9];
    const float* b1     = (const float*)d_in[10];
    const float* W2     = (const float*)d_in[11];
    const float* as2    = (const float*)d_in[12];
    const float* ad2    = (const float*)d_in[13];
    const float* b2     = (const float*)d_in[14];
    const float* pool_w = (const float*)d_in[15];
    const float* pool_b = (const float*)d_in[16];
    const float* WihF   = (const float*)d_in[17];
    const float* WhhF   = (const float*)d_in[18];
    const float* bihF   = (const float*)d_in[19];
    const float* bhhF   = (const float*)d_in[20];
    const float* WihR   = (const float*)d_in[21];
    const float* WhhR   = (const float*)d_in[22];
    const float* bihR   = (const float*)d_in[23];
    const float* bhhR   = (const float*)d_in[24];
    const float* w_mu   = (const float*)d_in[25];
    const float* b_mu   = (const float*)d_in[26];
    const float* w_lv   = (const float*)d_in[27];
    const float* b_lv   = (const float*)d_in[28];
    const float* w_pi   = (const float*)d_in[29];
    const float* b_pi   = (const float*)d_in[30];
    float* out = (float*)d_out;

    // workspace partition
    char* ws = (char*)d_ws;
    size_t off = 0;
    auto grab = [&](size_t bytes) { void* p = ws + off; off = (off + bytes + 255) & ~(size_t)255; return p; };
    _Float16* Af16   = (_Float16*)grab((size_t)NTOT * 256 * 2);  // current f16 activations
    _Float16* WT16   = (_Float16*)grab((size_t)256 * 256 * 2);   // current f16 weight, [N][K]
    float*    Hf32   = (float*)grab((size_t)NTOT * 256 * 4);     // GEMM output h
    float*    Xf32   = (float*)grab((size_t)NTOT * 256 * 4);     // layer output
    float*    AS     = (float*)grab((size_t)NTOT * HDS * 4);
    float*    AD     = (float*)grab((size_t)NTOT * HDS * 4);
    int*      csr_o  = (int*)grab((N_ + 1) * 4);
    int*      csr_l  = (int*)grab(EMAX * 4);
    _Float16* ge16   = (_Float16*)grab((size_t)G_ * HID * 2);
    _Float16* WihF16 = (_Float16*)grab((size_t)4 * RNN * HID * 2);
    _Float16* WhhF16 = (_Float16*)grab((size_t)4 * RNN * RNN * 2);
    _Float16* WihR16 = (_Float16*)grab((size_t)4 * RNN * HID * 2);
    _Float16* WhhR16 = (_Float16*)grab((size_t)4 * RNN * RNN * 2);
    float*    tf     = (float*)grab((size_t)B_ * 2 * RNN * 4);

    auto cvt = [&](const float* s, _Float16* d, int n) {
        f32_to_f16<<<(n + 255) / 256, 256, 0, stream>>>(s, d, n);
    };
    auto cvtT = [&](const float* W, _Float16* WT, int K, int N) {
        transpose_to_f16<<<(K * N + 255) / 256, 256, 0, stream>>>(W, WT, K, N);
    };

    build_csr<<<1, 64, 0, stream>>>(ei, csr_o, csr_l);

    const size_t agg_lds4 = (size_t)N_ * (HDS * HID) * 4;  // 64KB h-tile (H=4)
    const size_t agg_lds1 = (size_t)N_ * HID * 4;          // 16KB h-tile (H=1)

    // ---- GAT layer 0: [NTOT,32] x [32,256] -> agg (H=4,C=64) ----
    cvt(x, Af16, NTOT * F_);
    cvtT(W0, WT16, F_, 256);
    wmma_gemm_f16T<<<dim3(256 / 64, NTOT / 16), 32, 0, stream>>>(Af16, WT16, Hf32, NTOT, 256, F_);
    gat_scores<HDS, HID><<<(NTOT * HDS + 255) / 256, 256, 0, stream>>>(Hf32, as0, ad0, AS, AD);
    gat_aggregate<HDS, HID><<<G_, HDS * HID, agg_lds4, stream>>>(Hf32, AS, AD, csr_o, csr_l, b0, Xf32, Af16);

    // ---- GAT layer 1: [NTOT,256] x [256,256] ----
    cvtT(W1, WT16, 256, 256);
    wmma_gemm_f16T<<<dim3(256 / 64, NTOT / 16), 32, 0, stream>>>(Af16, WT16, Hf32, NTOT, 256, 256);
    gat_scores<HDS, HID><<<(NTOT * HDS + 255) / 256, 256, 0, stream>>>(Hf32, as1, ad1, AS, AD);
    gat_aggregate<HDS, HID><<<G_, HDS * HID, agg_lds4, stream>>>(Hf32, AS, AD, csr_o, csr_l, b1, Xf32, Af16);

    // ---- GAT layer 2 (heads=1): [NTOT,256] x [256,64] ----
    cvtT(W2, WT16, 256, HID);
    wmma_gemm_f16T<<<dim3(HID / 64, NTOT / 16), 32, 0, stream>>>(Af16, WT16, Hf32, NTOT, HID, 256);
    gat_scores<1, HID><<<(NTOT + 255) / 256, 256, 0, stream>>>(Hf32, as2, ad2, AS, AD);
    gat_aggregate<1, HID><<<G_, HID, agg_lds1, stream>>>(Hf32, AS, AD, csr_o, csr_l, b2, Xf32, Af16);

    // ---- attention pooling + mask -> ge16 [G, 64] ----
    pool_kernel<<<G_, N_, 0, stream>>>(Xf32, pool_w, pool_b, mask, ge16);

    // ---- bidirectional LSTM (weights already [N][K] in PyTorch layout) ----
    cvt(WihF, WihF16, 4 * RNN * HID);
    cvt(WhhF, WhhF16, 4 * RNN * RNN);
    cvt(WihR, WihR16, 4 * RNN * HID);
    cvt(WhhR, WhhR16, 4 * RNN * RNN);
    const size_t lstm_lds = (size_t)(4 * RNN * HID + 4 * RNN * RNN + 16 * RNN) * 2
                          + (size_t)(B_ * RNN + B_ * 4 * RNN) * 4;   // ~216KB < 320KB WGP LDS
    lstm_kernel<<<2, 512, lstm_lds, stream>>>(ge16, WihF16, WhhF16, bihF, bhhF,
                                              WihR16, WhhR16, bihR, bhhR, tf);

    // ---- GMM heads: mu(8x1024) ++ logvar(8x1024) ++ pi(8x32) ----
    head_kernel<<<(B_ * 1024 + 255) / 256, 256, 0, stream>>>(tf, w_mu, b_mu, out, 1024);
    head_kernel<<<(B_ * 1024 + 255) / 256, 256, 0, stream>>>(tf, w_lv, b_lv, out + B_ * 1024, 1024);
    head_kernel<<<(B_ * 32 + 255) / 256, 256, 0, stream>>>(tf, w_pi, b_pi, out + 2 * B_ * 1024, 32);
}
